// SimAMCNNMogrifierLSTM_45964740002336
// MI455X (gfx1250) — matrix operand order repaired and checked by reading, not compiled, not run on hardware
//
#include <hip/hip_runtime.h>

typedef __attribute__((ext_vector_type(16))) _Float16 v16h;
typedef __attribute__((ext_vector_type(8)))  _Float16 v8h;
typedef __attribute__((ext_vector_type(8)))  float    v8f;

#define SIMAM_LAM 0.1f

__device__ __forceinline__ float sigf(float x) { return 1.f / (1.f + __expf(-x)); }

// Load an A fragment (16-bit 16x32 WMMA layout): two contiguous 16B loads.
__device__ __forceinline__ v16h load_afrag(const _Float16* p) {
    v8h lo = *(const v8h*)p;
    v8h hi = *(const v8h*)(p + 16);
    return __builtin_shufflevector(lo, hi, 0, 1, 2, 3, 4, 5, 6, 7,
                                   8, 9, 10, 11, 12, 13, 14, 15);
}

#define WMMA(A, B, C) __builtin_amdgcn_wmma_f32_16x16x32_f16(false, A, false, B, (short)0, C, false, false)

// ---------------------------------------------------------------------------
// WMMA GEMM: out[m][n] = sum_k A[m][k] * Wpk[n][k] (+bias/epilogue).
// Block = 128 threads / 4 waves; block tile = 128(M) x 64(N); wave w owns rows
// [w*32, w*32+32) (2 M-tiles) and 4 N-subtiles -> 8 accumulators.
// K split across gridDim.z: if Pout != null, raw f32 partials are written to
// Pout[z][128][N] and a reduce kernel applies the epilogue.
// Direct-epilogue modes (gridDim.z == 1):
//   0: xm = 2*sigmoid(acc+b) * Xbuf[xm]      (q1/q2 mogrifier)
//   1: xm = 2*sigmoid(acc+b) * xt(h4,tstep)  (q0; xt gathered from conv output)
// Requires (K/32/gridDim.z) even (true for all shapes used here).
// ---------------------------------------------------------------------------
__global__ __launch_bounds__(128)
void gemm_wmma(const _Float16* __restrict__ A, int lda,
               const _Float16* __restrict__ Bpk,
               const float* __restrict__ bias,
               int N, int K, int mode,
               _Float16* __restrict__ Xbuf, int xstride, int destOff,
               const _Float16* __restrict__ h4, int tstep,
               float* __restrict__ Pout)
{
    const int lane = threadIdx.x & 31;
    const int wave = threadIdx.x >> 5;
    const int r  = lane & 15;
    const int kh = lane >> 4;
    const int colBase = blockIdx.x * 64;
    const int KC  = K >> 5;
    const int kcn = KC / gridDim.z;         // even by construction
    const int kc0 = kcn * blockIdx.z;
    const int row0 = wave * 32;

    const _Float16* a0 = A + (size_t)(row0 + r) * lda + (size_t)kc0 * 32 + (kh << 3);
    const _Float16* a1 = a0 + (size_t)16 * lda;
    const _Float16* bb = Bpk + ((size_t)(colBase >> 4) * KC + kc0) * 512 + lane * 16;
    const size_t bs = (size_t)KC * 512;     // packed stride per N-tile

    v8f c00 = {}, c01 = {}, c02 = {}, c03 = {};
    v8f c10 = {}, c11 = {}, c12 = {}, c13 = {};

    // software pipeline: stage "C" holds kc, stage "N" holds kc+1
    v16h aC0 = load_afrag(a0);
    v16h aC1 = load_afrag(a1);
    v16h bC0 = *(const v16h*)(bb);
    v16h bC1 = *(const v16h*)(bb + bs);
    v16h bC2 = *(const v16h*)(bb + 2 * bs);
    v16h bC3 = *(const v16h*)(bb + 3 * bs);
    v16h aN0, aN1, bN0, bN1, bN2, bN3;

    for (int kc = 0; kc < kcn; kc += 2) {
        // prefetch kc+1 while computing kc
        aN0 = load_afrag(a0 + (size_t)(kc + 1) * 32);
        aN1 = load_afrag(a1 + (size_t)(kc + 1) * 32);
        {
            const _Float16* bp = bb + (size_t)(kc + 1) * 512;
            bN0 = *(const v16h*)(bp);
            bN1 = *(const v16h*)(bp + bs);
            bN2 = *(const v16h*)(bp + 2 * bs);
            bN3 = *(const v16h*)(bp + 3 * bs);
        }
        c00 = WMMA(aC0, bC0, c00); c01 = WMMA(aC0, bC1, c01);
        c02 = WMMA(aC0, bC2, c02); c03 = WMMA(aC0, bC3, c03);
        c10 = WMMA(aC1, bC0, c10); c11 = WMMA(aC1, bC1, c11);
        c12 = WMMA(aC1, bC2, c12); c13 = WMMA(aC1, bC3, c13);
        if (kc + 2 < kcn) {
            // prefetch kc+2 while computing kc+1
            aC0 = load_afrag(a0 + (size_t)(kc + 2) * 32);
            aC1 = load_afrag(a1 + (size_t)(kc + 2) * 32);
            const _Float16* bp = bb + (size_t)(kc + 2) * 512;
            bC0 = *(const v16h*)(bp);
            bC1 = *(const v16h*)(bp + bs);
            bC2 = *(const v16h*)(bp + 2 * bs);
            bC3 = *(const v16h*)(bp + 3 * bs);
        }
        c00 = WMMA(aN0, bN0, c00); c01 = WMMA(aN0, bN1, c01);
        c02 = WMMA(aN0, bN2, c02); c03 = WMMA(aN0, bN3, c03);
        c10 = WMMA(aN1, bN0, c10); c11 = WMMA(aN1, bN1, c11);
        c12 = WMMA(aN1, bN2, c12); c13 = WMMA(aN1, bN3, c13);
    }

    v8f accs[2][4] = {{c00, c01, c02, c03}, {c10, c11, c12, c13}};
    if (Pout) {
        float* P = Pout + (size_t)blockIdx.z * 128 * N;
#pragma unroll
        for (int m = 0; m < 2; ++m)
#pragma unroll
            for (int j = 0; j < 4; ++j) {
                int col = colBase + j * 16 + r;
#pragma unroll
                for (int e = 0; e < 8; ++e) {
                    int row = row0 + m * 16 + (kh << 3) + e;
                    P[(size_t)row * N + col] = accs[m][j][e];
                }
            }
        return;
    }
#pragma unroll
    for (int m = 0; m < 2; ++m)
#pragma unroll
        for (int j = 0; j < 4; ++j) {
            int col = colBase + j * 16 + r;
            float bsv = bias[col];
#pragma unroll
            for (int e = 0; e < 8; ++e) {
                int row = row0 + m * 16 + (kh << 3) + e;
                float v = accs[m][j][e] + bsv;
                float sc = 2.f * sigf(v);
                float oldv;
                if (mode == 1) {
                    // xl[b,t',k] = h4[b, t'>>1, (t'&1)*64 + (k>>6), k&63]
                    int cch = tstep >> 1;
                    int tt  = ((tstep & 1) << 6) + (col >> 6);
                    int dd  = col & 63;
                    oldv = (float)h4[(((size_t)row * 64 + cch) * 128 + tt) * 64 + dd];
                } else {
                    oldv = (float)Xbuf[(size_t)row * xstride + destOff + col];
                }
                Xbuf[(size_t)row * xstride + destOff + col] = (_Float16)(sc * oldv);
            }
        }
}

// Split-K reduce + mogrifier-h epilogue: hm = 2*sig(sum_z P + bias) * hm
__global__ __launch_bounds__(256)
void reduce_moghm_k(const float* __restrict__ P, int Z, int N,
                    const float* __restrict__ bias,
                    _Float16* __restrict__ Xbuf, int xstride, int destOff)
{
    int i = blockIdx.x * 256 + threadIdx.x;   // 128*N threads
    int row = i / N, col = i - row * N;
    float v = bias[col];
    for (int z = 0; z < Z; ++z) v += P[((size_t)z * 128 + row) * N + col];
    float sc = 2.f * sigf(v);
    size_t o = (size_t)row * xstride + destOff + col;
    Xbuf[o] = (_Float16)(sc * (float)Xbuf[o]);
}

// Split-K reduce of gate partials fused with LSTM cell update.
__global__ __launch_bounds__(256)
void gatecell_k(const float* __restrict__ P, int Z,
                const float* __restrict__ gbias,
                float* __restrict__ C, _Float16* __restrict__ Xbuf)
{
    int i = blockIdx.x * 256 + threadIdx.x;   // 128*256
    int b = i >> 8, n = i & 255;
    float g4[4];
#pragma unroll
    for (int gi = 0; gi < 4; ++gi) {
        int col = gi * 256 + n;
        float v = gbias[col];
        for (int z = 0; z < Z; ++z) v += P[((size_t)z * 128 + b) * 1024 + col];
        g4[gi] = v;
    }
    float ig = sigf(g4[0]), fg = sigf(g4[1]);
    float gg = tanhf(g4[2]), og = sigf(g4[3]);
    float c = fg * C[i] + ig * gg;
    C[i] = c;
    Xbuf[(size_t)b * 4352 + 4096 + n] = (_Float16)(og * tanhf(c));
}

// ------------------------------- CNN part ----------------------------------

__global__ void fold_bn_k(const float* __restrict__ w, const float* __restrict__ bc,
                          const float* __restrict__ g, const float* __restrict__ bb,
                          const float* __restrict__ m, const float* __restrict__ v,
                          _Float16* __restrict__ wf, float* __restrict__ bf,
                          int total, int per_co)
{
    int idx = blockIdx.x * 256 + threadIdx.x;
    if (idx >= total) return;
    int c = idx / per_co;
    float s = g[c] * rsqrtf(v[c] + 1e-5f);
    wf[idx] = (_Float16)(w[idx] * s);
    if (idx - c * per_co == 0) bf[c] = (bc[c] - m[c]) * s + bb[c];
}

// Zero-fill f16 buffer (count multiple of 8).
__global__ void zerof16_k(_Float16* __restrict__ p, size_t n)
{
    size_t i = ((size_t)blockIdx.x * 256 + threadIdx.x) * 8;
    if (i < n) { v8h z = {}; *(v8h*)(p + i) = z; }
}

// conv1a: 1x1 1->32 + BN + ReLU, writes padded channel-innermost P1 interior.
__global__ __launch_bounds__(256)
void conv1a_k(const float* __restrict__ x, const _Float16* __restrict__ w,
              const float* __restrict__ bias, _Float16* __restrict__ P1)
{
    int s = blockIdx.x * 256 + threadIdx.x;   // t*64+d
    int b = blockIdx.z;
    int t = s >> 6, d = s & 63;
    float xv = x[(size_t)b * 8192 + s];
    _Float16* o = P1 + (((size_t)b * 130 + t + 1) * 66 + d + 1) * 32;
#pragma unroll
    for (int co = 0; co < 32; ++co) {
        float v = xv * (float)w[co] + bias[co];
        o[co] = (_Float16)(v > 0.f ? v : 0.f);
    }
}

// conv2a: 1x1 32->64 + BN + ReLU, S1 (b,s,32) -> padded P2 interior.
__global__ __launch_bounds__(256)
void conv2a_k(const _Float16* __restrict__ S1, const _Float16* __restrict__ w,
              const float* __restrict__ bias, _Float16* __restrict__ P2)
{
    int s = blockIdx.x * 256 + threadIdx.x;
    int b = blockIdx.z;
    int t = s >> 6, d = s & 63;
    float iv[32];
    const _Float16* ip = S1 + ((size_t)b * 8192 + s) * 32;
#pragma unroll
    for (int ci = 0; ci < 32; ++ci) iv[ci] = (float)ip[ci];
    _Float16* o = P2 + (((size_t)b * 130 + t + 1) * 66 + d + 1) * 64;
    for (int co = 0; co < 64; ++co) {
        float acc = bias[co];
        const _Float16* wp = w + co * 32;
#pragma unroll
        for (int ci = 0; ci < 32; ++ci) acc += iv[ci] * (float)wp[ci];
        o[co] = (_Float16)(acc > 0.f ? acc : 0.f);
    }
}

// Pack folded conv weights (co,ci,3,3) into per-tap WMMA B-fragment layout.
__global__ void pack_conv_k(const _Float16* __restrict__ wf, _Float16* __restrict__ out,
                            int CO, int CI)
{
    int idx = blockIdx.x * 256 + threadIdx.x;   // 9*CO*CI
    if (idx >= 9 * CO * CI) return;
    int tap = idx / (CO * CI);
    int rem = idx - tap * CO * CI;
    int n = rem / CI;
    int k = rem - n * CI;
    int KC = CI >> 5;
    int krem = k & 31;
    size_t o = (size_t)tap * CO * CI
             + (((size_t)(n >> 4) * KC + (k >> 5)) * 32 + (((krem >> 3) & 1) * 16 + (n & 15))) * 16
             + ((krem >> 4) * 8 + (krem & 7));
    out[o] = wf[((size_t)n * CI + k) * 9 + tap];
}

// 3x3 conv as 9-tap implicit GEMM on WMMA.  Padded input (b,130,66,CI), no
// boundary branches.  Block = 4 waves; wave w = d-tile w; grid (t=128,1,b=128).
// CO==32 -> write S1 (b,s,32); CO==64 -> write h4 (b,c,t,d) with v8h stores.
template<int CI, int CO>
__global__ __launch_bounds__(128)
void conv3x3_wmma(const _Float16* __restrict__ P, const _Float16* __restrict__ Wp,
                  const float* __restrict__ bias, _Float16* __restrict__ outS,
                  _Float16* __restrict__ outH)
{
    constexpr int KC = CI / 32;
    constexpr int NT = CO / 16;
    const int lane = threadIdx.x & 31, wave = threadIdx.x >> 5;
    const int r = lane & 15, kh = lane >> 4;
    const int t = blockIdx.x, b = blockIdx.z;
    const int d0 = wave * 16;
    const _Float16* Pb = P + (size_t)b * 130 * 66 * CI;
    const _Float16* Wl = Wp + lane * 16;

    v8f acc[NT];
#pragma unroll
    for (int j = 0; j < NT; ++j) acc[j] = (v8f){};

#pragma unroll
    for (int tap = 0; tap < 9; ++tap) {
        const int kt = tap / 3, kd = tap % 3;
        const _Float16* arow =
            Pb + (((size_t)(t + kt)) * 66 + (d0 + r + kd)) * CI + (kh << 3);
        const _Float16* bt = Wl + (size_t)tap * CO * CI;
#pragma unroll
        for (int kc = 0; kc < KC; ++kc) {
            v16h a = load_afrag(arow + kc * 32);
#pragma unroll
            for (int j = 0; j < NT; ++j) {
                v16h bf = *(const v16h*)(bt + ((size_t)j * KC + kc) * 512);
                acc[j] = WMMA(a, bf, acc[j]);
            }
        }
    }
    if (CO == 32) {
#pragma unroll
        for (int j = 0; j < NT; ++j) {
            int co = j * 16 + r;
            float bsv = bias[co];
#pragma unroll
            for (int e = 0; e < 8; ++e) {
                int d = d0 + (kh << 3) + e;
                float v = acc[j][e] + bsv;
                outS[((size_t)b * 8192 + t * 64 + d) * 32 + co] =
                    (_Float16)(v > 0.f ? v : 0.f);
            }
        }
    } else {
#pragma unroll
        for (int j = 0; j < NT; ++j) {
            int co = j * 16 + r;
            float bsv = bias[co];
            v8h st;
#pragma unroll
            for (int e = 0; e < 8; ++e) {
                float v = acc[j][e] + bsv;
                st[e] = (_Float16)(v > 0.f ? v : 0.f);
            }
            *(v8h*)(outH + (((size_t)b * 64 + co) * 128 + t) * 64 + d0 + (kh << 3)) = st;
        }
    }
}

// SimAM in-place; stride==1: planes contiguous (b,c,t,d); else (b,s,C) layout.
__global__ __launch_bounds__(256)
void simam_k(_Float16* __restrict__ buf, int C, int stride)
{
    __shared__ float s1[256], s2[256];
    int blk = blockIdx.x, b = blk / C, c = blk - b * C;
    _Float16* p = (stride == 1) ? buf + (size_t)blk * 8192
                                : buf + (size_t)b * 8192 * C + c;
    int tid = threadIdx.x;
    float s = 0.f, ss = 0.f;
    for (int i = tid; i < 8192; i += 256) {
        float v = (float)p[(size_t)i * stride];
        s += v; ss += v * v;
    }
    s1[tid] = s; s2[tid] = ss;
    __syncthreads();
    for (int off = 128; off > 0; off >>= 1) {
        if (tid < off) { s1[tid] += s1[tid + off]; s2[tid] += s2[tid + off]; }
        __syncthreads();
    }
    float S = s1[0], SS = s2[0];
    const float n1 = 8191.f, n2 = 8190.f;
    for (int i = tid; i < 8192; i += 256) {
        float x = (float)p[(size_t)i * stride];
        float mu = (S - x) / n1;
        float var = (SS - x * x - n1 * mu * mu) / n2;
        float einv = ((x - mu) * (x - mu) + 2.f * var + 2.f * SIMAM_LAM)
                   / (4.f * (var + SIMAM_LAM));
        p[(size_t)i * stride] = (_Float16)(x * sigf(einv));
    }
}

// ----------------------------- LSTM helpers --------------------------------

__global__ void pack_w_k(const float* __restrict__ W1, int K1,
                         const float* __restrict__ W2, int K2,
                         _Float16* __restrict__ out, int N, int K)
{
    size_t idx = (size_t)blockIdx.x * 256 + threadIdx.x;
    if (idx >= (size_t)N * K) return;
    int n = (int)(idx / K), k = (int)(idx % K);
    float v = (k < K1) ? W1[(size_t)n * K1 + k] : W2[(size_t)n * K2 + (k - K1)];
    int krem = k & 31;
    size_t o = (((size_t)(n >> 4) * (K >> 5) + (k >> 5)) * 32
                + (((krem >> 3) & 1) * 16 + (n & 15))) * 16 + ((krem >> 4) * 8 + (krem & 7));
    out[o] = (_Float16)v;
}

__global__ void gbias_k(const float* __restrict__ a, const float* __restrict__ b,
                        float* __restrict__ o)
{
    int i = blockIdx.x * 256 + threadIdx.x;
    if (i < 1024) o[i] = a[i] + b[i];
}

__global__ void init_state_k(float* __restrict__ C, _Float16* __restrict__ Xbuf)
{
    int i = blockIdx.x * 256 + threadIdx.x;   // 128*256
    C[i] = 0.f;
    int b = i >> 8, n = i & 255;
    Xbuf[(size_t)b * 4352 + 4096 + n] = (_Float16)0.f;
}

__global__ void fc_k(const _Float16* __restrict__ Xbuf, const float* __restrict__ fw,
                     const float* __restrict__ fb, float* __restrict__ out)
{
    int i = blockIdx.x * 256 + threadIdx.x;
    if (i >= 640) return;
    int b = i / 5, j = i % 5;
    float acc = fb[j];
    const _Float16* h = Xbuf + (size_t)b * 4352 + 4096;
    for (int n = 0; n < 256; ++n) acc += (float)h[n] * fw[j * 256 + n];
    out[i] = acc;
}

// ---------------------------------------------------------------------------

extern "C" void kernel_launch(void* const* d_in, const int* in_sizes, int n_in,
                              void* d_out, int out_size, void* d_ws, size_t ws_size,
                              hipStream_t stream)
{
    const float* x     = (const float*)d_in[0];
    const float* c1a_w = (const float*)d_in[1];
    const float* c1a_b = (const float*)d_in[2];
    const float* g1a   = (const float*)d_in[3];
    const float* b1a   = (const float*)d_in[4];
    const float* m1a   = (const float*)d_in[5];
    const float* v1a   = (const float*)d_in[6];
    const float* c1b_w = (const float*)d_in[7];
    const float* c1b_b = (const float*)d_in[8];
    const float* g1b   = (const float*)d_in[9];
    const float* b1b   = (const float*)d_in[10];
    const float* m1b   = (const float*)d_in[11];
    const float* v1b   = (const float*)d_in[12];
    const float* c2a_w = (const float*)d_in[13];
    const float* c2a_b = (const float*)d_in[14];
    const float* g2a   = (const float*)d_in[15];
    const float* b2a   = (const float*)d_in[16];
    const float* m2a   = (const float*)d_in[17];
    const float* v2a   = (const float*)d_in[18];
    const float* c2b_w = (const float*)d_in[19];
    const float* c2b_b = (const float*)d_in[20];
    const float* g2b   = (const float*)d_in[21];
    const float* b2b   = (const float*)d_in[22];
    const float* m2b   = (const float*)d_in[23];
    const float* v2b   = (const float*)d_in[24];
    const float* qw[3] = {(const float*)d_in[25], (const float*)d_in[29], (const float*)d_in[33]};
    const float* qb[3] = {(const float*)d_in[26], (const float*)d_in[30], (const float*)d_in[34]};
    const float* rw[3] = {(const float*)d_in[27], (const float*)d_in[31], (const float*)d_in[35]};
    const float* rb[3] = {(const float*)d_in[28], (const float*)d_in[32], (const float*)d_in[36]};
    const float* wih   = (const float*)d_in[37];
    const float* whh   = (const float*)d_in[38];
    const float* bih   = (const float*)d_in[39];
    const float* bhh   = (const float*)d_in[40];
    const float* fc_w  = (const float*)d_in[41];
    const float* fc_b  = (const float*)d_in[42];

    char* ws = (char*)d_ws;
    size_t off = 0;
    auto carve = [&](size_t bytes) -> char* {
        char* p = ws + off;
        off += (bytes + 255) & ~(size_t)255;
        return p;
    };
    const size_t nP1 = (size_t)128 * 130 * 66 * 32;   // padded conv1 input act.
    const size_t nP2 = (size_t)128 * 130 * 66 * 64;
    _Float16* P1   = (_Float16*)carve(nP1 * 2);
    _Float16* S1   = (_Float16*)carve((size_t)128 * 8192 * 32 * 2);
    _Float16* P2   = (_Float16*)carve(nP2 * 2);
    _Float16* h4   = (_Float16*)carve((size_t)128 * 64 * 8192 * 2);
    _Float16 *pq[3], *pr[3];
    for (int i = 0; i < 3; ++i) pq[i] = (_Float16*)carve((size_t)4096 * 256 * 2);
    for (int i = 0; i < 3; ++i) pr[i] = (_Float16*)carve((size_t)256 * 4096 * 2);
    _Float16* pw    = (_Float16*)carve((size_t)1024 * 4352 * 2);
    _Float16* Xbuf  = (_Float16*)carve((size_t)128 * 4352 * 2);
    float*    Cbuf  = (float*)carve((size_t)128 * 256 * 4);
    float*    Pr    = (float*)carve((size_t)8 * 128 * 256 * 4);
    float*    Pg    = (float*)carve((size_t)4 * 128 * 1024 * 4);
    float*    gbias = (float*)carve(1024 * 4);
    _Float16* wf1a  = (_Float16*)carve(32 * 2);
    float*    bf1a  = (float*)carve(32 * 4);
    _Float16* wf1b  = (_Float16*)carve(32 * 32 * 9 * 2);
    float*    bf1b  = (float*)carve(32 * 4);
    _Float16* wf2a  = (_Float16*)carve(64 * 32 * 2);
    float*    bf2a  = (float*)carve(64 * 4);
    _Float16* wf2b  = (_Float16*)carve(64 * 64 * 9 * 2);
    float*    bf2b  = (float*)carve(64 * 4);
    _Float16* pc1b  = (_Float16*)carve((size_t)9 * 32 * 32 * 2);
    _Float16* pc2b  = (_Float16*)carve((size_t)9 * 64 * 64 * 2);

    // --- prep: BN fold, weight packing, padded-buffer zeroing ---
    fold_bn_k<<<1, 256, 0, stream>>>(c1a_w, c1a_b, g1a, b1a, m1a, v1a, wf1a, bf1a, 32, 1);
    fold_bn_k<<<36, 256, 0, stream>>>(c1b_w, c1b_b, g1b, b1b, m1b, v1b, wf1b, bf1b, 9216, 288);
    fold_bn_k<<<8, 256, 0, stream>>>(c2a_w, c2a_b, g2a, b2a, m2a, v2a, wf2a, bf2a, 2048, 32);
    fold_bn_k<<<144, 256, 0, stream>>>(c2b_w, c2b_b, g2b, b2b, m2b, v2b, wf2b, bf2b, 36864, 576);
    pack_conv_k<<<36, 256, 0, stream>>>(wf1b, pc1b, 32, 32);
    pack_conv_k<<<144, 256, 0, stream>>>(wf2b, pc2b, 64, 64);
    for (int i = 0; i < 3; ++i) {
        pack_w_k<<<4096, 256, 0, stream>>>(qw[i], 256, nullptr, 0, pq[i], 4096, 256);
        pack_w_k<<<4096, 256, 0, stream>>>(rw[i], 4096, nullptr, 0, pr[i], 256, 4096);
    }
    pack_w_k<<<17408, 256, 0, stream>>>(wih, 4096, whh, 256, pw, 1024, 4352);
    gbias_k<<<4, 256, 0, stream>>>(bih, bhh, gbias);
    zerof16_k<<<(unsigned)((nP1 / 8 + 255) / 256), 256, 0, stream>>>(P1, nP1);
    zerof16_k<<<(unsigned)((nP2 / 8 + 255) / 256), 256, 0, stream>>>(P2, nP2);

    // --- CNN (convs on WMMA) ---
    conv1a_k<<<dim3(32, 1, 128), 256, 0, stream>>>(x, wf1a, bf1a, P1);
    conv3x3_wmma<32, 32><<<dim3(128, 1, 128), 128, 0, stream>>>(P1, pc1b, bf1b, S1, nullptr);
    simam_k<<<128 * 32, 256, 0, stream>>>(S1, 32, 32);
    conv2a_k<<<dim3(32, 1, 128), 256, 0, stream>>>(S1, wf2a, bf2a, P2);
    conv3x3_wmma<64, 64><<<dim3(128, 1, 128), 128, 0, stream>>>(P2, pc2b, bf2b, nullptr, h4);
    simam_k<<<128 * 64, 256, 0, stream>>>(h4, 64, 1);

    // --- Mogrifier LSTM ---
    init_state_k<<<128, 256, 0, stream>>>(Cbuf, Xbuf);
    _Float16* Xh = Xbuf + 4096;   // hm view (row stride 4352)
    dim3 blk(128);
    for (int t = 0; t < 128; ++t) {
        gemm_wmma<<<dim3(64, 1, 1), blk, 0, stream>>>(Xh, 4352, pq[0], qb[0], 4096, 256, 1,
                                                      Xbuf, 4352, 0, h4, t, nullptr);
        gemm_wmma<<<dim3(4, 1, 8), blk, 0, stream>>>(Xbuf, 4352, pr[0], rb[0], 256, 4096, 0,
                                                     nullptr, 0, 0, nullptr, 0, Pr);
        reduce_moghm_k<<<128, 256, 0, stream>>>(Pr, 8, 256, rb[0], Xbuf, 4352, 4096);
        gemm_wmma<<<dim3(64, 1, 1), blk, 0, stream>>>(Xh, 4352, pq[1], qb[1], 4096, 256, 0,
                                                      Xbuf, 4352, 0, nullptr, 0, nullptr);
        gemm_wmma<<<dim3(4, 1, 8), blk, 0, stream>>>(Xbuf, 4352, pr[1], rb[1], 256, 4096, 0,
                                                     nullptr, 0, 0, nullptr, 0, Pr);
        reduce_moghm_k<<<128, 256, 0, stream>>>(Pr, 8, 256, rb[1], Xbuf, 4352, 4096);
        gemm_wmma<<<dim3(64, 1, 1), blk, 0, stream>>>(Xh, 4352, pq[2], qb[2], 4096, 256, 0,
                                                      Xbuf, 4352, 0, nullptr, 0, nullptr);
        gemm_wmma<<<dim3(4, 1, 8), blk, 0, stream>>>(Xbuf, 4352, pr[2], rb[2], 256, 4096, 0,
                                                     nullptr, 0, 0, nullptr, 0, Pr);
        reduce_moghm_k<<<128, 256, 0, stream>>>(Pr, 8, 256, rb[2], Xbuf, 4352, 4096);
        // gates = [xm|hm] @ [wih|whh]^T, split-K=4, reduce fused with cell
        gemm_wmma<<<dim3(16, 1, 4), blk, 0, stream>>>(Xbuf, 4352, pw, gbias, 1024, 4352, 0,
                                                      nullptr, 0, 0, nullptr, 0, Pg);
        gatecell_k<<<128, 256, 0, stream>>>(Pg, 4, gbias, Cbuf, Xbuf);
    }

    fc_k<<<3, 256, 0, stream>>>(Xbuf, fc_w, fc_b, (float*)d_out);
}
